// NetVLAD_22308060135850
// MI455X (gfx1250) — matrix operand order repaired and checked by reading, not compiled, not run on hardware
//
#include <hip/hip_runtime.h>
#include <hip/hip_bf16.h>

typedef __attribute__((ext_vector_type(16))) __bf16 v16bf;
typedef __attribute__((ext_vector_type(8)))  __bf16 v8bf;
typedef __attribute__((ext_vector_type(8)))  float  v8f;

#define NB   64      // batch
#define CC   512     // channels
#define KK   64      // clusters
#define HWP  900     // pixels (30*30)
#define PP   928     // padded pixels: 29*32 = 58*16
#define EPSN 1e-12f

// ---------------- Phase 0: conv_w fp32 -> bf16 ----------------
__global__ void k_cvt_w(const float* __restrict__ w, __bf16* __restrict__ wbf, int n) {
    int i = blockIdx.x * blockDim.x + threadIdx.x;
    if (i < n) wbf[i] = (__bf16)w[i];
}

// -- Phase 1: per-pixel L2 norm scale; write XT[n][p][c] (bf16, c-contig) + sc[n][p] --
__global__ void k_norm(const float* __restrict__ x, __bf16* __restrict__ XT,
                       float* __restrict__ scs) {
    int p = blockIdx.x * blockDim.x + threadIdx.x;   // 0..1023
    int n = blockIdx.y;
    if (p >= PP) return;
    __bf16* row = XT + ((size_t)n * PP + p) * CC;
    if (p >= HWP) {                                   // zero padding rows
        float4 z4 = make_float4(0.f, 0.f, 0.f, 0.f);
        #pragma unroll 4
        for (int j = 0; j < CC / 8; ++j) ((float4*)row)[j] = z4;
        scs[(size_t)n * PP + p] = 0.f;
        return;
    }
    const float* xp = x + (size_t)n * CC * HWP + p;   // x[n][c][p], stride HWP over c
    float ss = 0.f;
    for (int c = 0; c < CC; ++c) { float v = xp[(size_t)c * HWP]; ss += v * v; }
    float sc = 1.0f / fmaxf(sqrtf(ss), EPSN);
    scs[(size_t)n * PP + p] = sc;
    for (int c = 0; c < CC; c += 8) {
        v8bf o;
        #pragma unroll
        for (int t = 0; t < 8; ++t) o[t] = (__bf16)(xp[(size_t)(c + t) * HWP] * sc);
        *(v8bf*)(row + c) = o;
    }
}

// ------ Phase 2: logits = W(64x512) @ X(512xP), softmax over K, store A bf16 ------
__global__ void k_logits(const __bf16* __restrict__ Wbf, const __bf16* __restrict__ XT,
                         __bf16* __restrict__ Abf) {
    int n    = blockIdx.y;
    int wave = threadIdx.x >> 5;
    int lane = threadIdx.x & 31;
    int pt   = blockIdx.x * 8 + wave;                 // pixel tile (16 pixels)
    if (pt >= PP / 16) return;                        // wave-uniform
    int h    = lane >> 4;                             // half select
    int r15  = lane & 15;
    int pcol = pt * 16 + r15;                         // this lane's pixel column

    v8f acc[4];
    #pragma unroll
    for (int i = 0; i < 4; ++i)
        #pragma unroll
        for (int t = 0; t < 8; ++t) acc[i][t] = 0.f;

    const __bf16* xrow = XT + ((size_t)n * PP + pcol) * CC;
    for (int c0 = 0; c0 < CC; c0 += 32) {
        // B fragment: element t = X[c0+16h+t][pcol] -> contiguous in XT
        v16bf b = *(const v16bf*)(xrow + c0 + 16 * h);
        #pragma unroll
        for (int mt = 0; mt < 4; ++mt) {
            const __bf16* wr = Wbf + (size_t)(mt * 16 + r15) * CC + c0 + 16 * h;
            v8bf lo = *(const v8bf*)wr;
            v8bf hi = *(const v8bf*)(wr + 8);
            v16bf a;
            #pragma unroll
            for (int t = 0; t < 8; ++t) { a[t] = lo[t]; a[8 + t] = hi[t]; }
            acc[mt] = __builtin_amdgcn_wmma_f32_16x16x32_bf16(
                false, a, false, b, (short)0, acc[mt], false, false);
        }
    }

    // softmax over 64 clusters for pixel pcol; lanes (L, L^16) jointly hold all 64
    float mx = -3.4e38f;
    #pragma unroll
    for (int mt = 0; mt < 4; ++mt)
        #pragma unroll
        for (int r = 0; r < 8; ++r) mx = fmaxf(mx, acc[mt][r]);
    mx = fmaxf(mx, __shfl_xor(mx, 16, 32));
    float e[32];
    float s = 0.f;
    #pragma unroll
    for (int mt = 0; mt < 4; ++mt)
        #pragma unroll
        for (int r = 0; r < 8; ++r) { float v = __expf(acc[mt][r] - mx); e[mt * 8 + r] = v; s += v; }
    s += __shfl_xor(s, 16, 32);
    float inv = 1.0f / s;
    bool valid = (pcol < HWP);
    #pragma unroll
    for (int mt = 0; mt < 4; ++mt)
        #pragma unroll
        for (int r = 0; r < 8; ++r) {
            int m = mt * 16 + 8 * h + r;
            Abf[((size_t)n * KK + m) * PP + pcol] = (__bf16)(valid ? e[mt * 8 + r] * inv : 0.f);
        }
}

// ---------------- Phase 2b: asum[n][k] = sum_p A[n][k][p] ----------------
__global__ void k_asum(const __bf16* __restrict__ Abf, float* __restrict__ asum) {
    int i = blockIdx.x * blockDim.x + threadIdx.x;    // n*KK + k
    if (i >= NB * KK) return;
    const __bf16* row = Abf + (size_t)i * PP;
    float s = 0.f;
    for (int p = 0; p < PP; ++p) s += (float)row[p];
    asum[i] = s;
}

// ------ Phase 3: agg = A(64xP) @ (sc.*x)^T(Px512); vlad = agg - asum*centroids ------
// B fragment built from raw fp32 x[n][c][p] (p-contiguous) scaled by sc[p] on the fly.
__global__ void k_vlad(const __bf16* __restrict__ Abf, const float* __restrict__ x,
                       const float* __restrict__ scs, const float* __restrict__ asum,
                       const float* __restrict__ centroids, float* __restrict__ out) {
    int n    = blockIdx.x;
    int wave = threadIdx.x >> 5;
    int lane = threadIdx.x & 31;
    int h    = lane >> 4;
    int r15  = lane & 15;
    const float* scp = scs + (size_t)n * PP;

    for (int ct = wave; ct < CC / 16; ct += 8) {      // channel tiles, wave-uniform
        int cbase = ct * 16;
        const float* xcol = x + ((size_t)n * CC + cbase + r15) * HWP; // row c = cbase+r15
        v8f acc[4];
        #pragma unroll
        for (int i = 0; i < 4; ++i)
            #pragma unroll
            for (int t = 0; t < 8; ++t) acc[i][t] = 0.f;

        for (int p0 = 0; p0 < PP; p0 += 32) {
            int pb = p0 + 16 * h;
            // B fragment: element t = xn[cbase+r15][pb+t] = x[..][pb+t]*sc[pb+t]
            v16bf b;
            if (p0 < 896) {                            // uniform branch: fully in-range
                float4 x0 = *(const float4*)(xcol + pb);
                float4 x1 = *(const float4*)(xcol + pb + 4);
                float4 x2 = *(const float4*)(xcol + pb + 8);
                float4 x3 = *(const float4*)(xcol + pb + 12);
                float4 s0 = *(const float4*)(scp + pb);
                float4 s1 = *(const float4*)(scp + pb + 4);
                float4 s2 = *(const float4*)(scp + pb + 8);
                float4 s3 = *(const float4*)(scp + pb + 12);
                b[0]  = (__bf16)(x0.x * s0.x); b[1]  = (__bf16)(x0.y * s0.y);
                b[2]  = (__bf16)(x0.z * s0.z); b[3]  = (__bf16)(x0.w * s0.w);
                b[4]  = (__bf16)(x1.x * s1.x); b[5]  = (__bf16)(x1.y * s1.y);
                b[6]  = (__bf16)(x1.z * s1.z); b[7]  = (__bf16)(x1.w * s1.w);
                b[8]  = (__bf16)(x2.x * s2.x); b[9]  = (__bf16)(x2.y * s2.y);
                b[10] = (__bf16)(x2.z * s2.z); b[11] = (__bf16)(x2.w * s2.w);
                b[12] = (__bf16)(x3.x * s3.x); b[13] = (__bf16)(x3.y * s3.y);
                b[14] = (__bf16)(x3.z * s3.z); b[15] = (__bf16)(x3.w * s3.w);
            } else {                                   // last chunk: mask p >= 900
                #pragma unroll
                for (int t = 0; t < 16; ++t) {
                    int p = pb + t;
                    float v = (p < HWP) ? xcol[p] * scp[p] : 0.f;
                    b[t] = (__bf16)v;
                }
            }
            #pragma unroll
            for (int mt = 0; mt < 4; ++mt) {
                const __bf16* ar = Abf + ((size_t)n * KK + mt * 16 + r15) * PP + p0 + 16 * h;
                v8bf lo = *(const v8bf*)ar;
                v8bf hi = *(const v8bf*)(ar + 8);
                v16bf a;
                #pragma unroll
                for (int t = 0; t < 8; ++t) { a[t] = lo[t]; a[8 + t] = hi[t]; }
                acc[mt] = __builtin_amdgcn_wmma_f32_16x16x32_bf16(
                    false, a, false, b, (short)0, acc[mt], false, false);
            }
        }

        int c = cbase + r15;
        #pragma unroll
        for (int mt = 0; mt < 4; ++mt)
            #pragma unroll
            for (int r = 0; r < 8; ++r) {
                int m = mt * 16 + 8 * h + r;
                out[((size_t)n * KK + m) * CC + c] =
                    acc[mt][r] - asum[n * KK + m] * centroids[(size_t)m * CC + c];
            }
    }
}

extern "C" void kernel_launch(void* const* d_in, const int* in_sizes, int n_in,
                              void* d_out, int out_size, void* d_ws, size_t ws_size,
                              hipStream_t stream) {
    const float* x         = (const float*)d_in[0];   // [64,512,30,30]
    const float* conv_w    = (const float*)d_in[1];   // [64,512]
    const float* centroids = (const float*)d_in[2];   // [64,512]
    float* out = (float*)d_out;                       // [64,64,512]

    // workspace layout
    __bf16* Wbf = (__bf16*)d_ws;                      // 64*512
    __bf16* XT  = Wbf + (size_t)KK * CC;              // N*PP*CC  (p-major, bf16)
    __bf16* Abf = XT + (size_t)NB * PP * CC;          // N*KK*PP  (softmax weights, bf16)
    float*  asum = (float*)(Abf + (size_t)NB * KK * PP); // N*KK fp32
    float*  scs  = asum + (size_t)NB * KK;            // N*PP fp32 (norm scales)

    k_cvt_w <<<dim3((KK * CC + 255) / 256), 256, 0, stream>>>(conv_w, Wbf, KK * CC);
    k_norm  <<<dim3((PP + 255) / 256, NB),  256, 0, stream>>>(x, XT, scs);
    k_logits<<<dim3(8, NB),                 256, 0, stream>>>(Wbf, XT, Abf);
    k_asum  <<<dim3((NB * KK + 255) / 256), 256, 0, stream>>>(Abf, asum);
    k_vlad  <<<dim3(NB),                    256, 0, stream>>>(Abf, x, scs, asum, centroids, out);
}
